// KozyraTwoStageModel_48473000902921
// MI455X (gfx1250) — matrix operand order repaired and checked by reading, not compiled, not run on hardware
//
#include <hip/hip_runtime.h>
#include <hip/hip_bf16.h>

typedef __attribute__((ext_vector_type(16))) __bf16 v16bf;
typedef __attribute__((ext_vector_type(8)))  __bf16 v8bf;
typedef __attribute__((ext_vector_type(8)))  float  v8f;

#define TSTEPS   128
#define SD       16
#define HID      50
#define G4       200           // 4*HID
#define NPAD     208           // 13 tiles of 16
#define NT       13
#define K0PAD    96            // layer0: [x(16), delta(1), h0(50)] = 67 -> 96  (3 k-steps)
#define K1PAD    128           // layer1: [h0(50), h1(50)] = 100 -> 128        (4 k-steps)
#define NKT0     3
#define NKT1     4
#define ROWS     16            // batch rows per workgroup (WMMA M)
#define NTHREADS 128           // 4 waves (wave32)

// ---------------- LDS layout (bytes, all 16B aligned) ----------------
#define OFF_W0F   0
#define SZ_W0F    (NKT0*NT*512*2)      // 39936  : B-fragments, layer0
#define OFF_W1F   (OFF_W0F + SZ_W0F)
#define SZ_W1F    (NKT1*NT*512*2)      // 53248  : B-fragments, layer1
#define OFF_GATES (OFF_W1F + SZ_W1F)
#define SZ_GATES  (ROWS*NPAD*4)        // 13312
#define OFF_XH0   (OFF_GATES + SZ_GATES)
#define SZ_XH0    (ROWS*K0PAD*2)       // 3072   : bf16 [x|delta|h0] rows
#define OFF_XH1   (OFF_XH0 + SZ_XH0)
#define SZ_XH1    (ROWS*K1PAD*2)       // 4096   : bf16 [h0|h1] rows
#define OFF_H1F   (OFF_XH1 + SZ_XH1)
#define SZ_H1F    (ROWS*52*4)          // 3328   : f32 h1 for the FC head
#define OFF_S     (OFF_H1F + SZ_H1F)
#define SZ_S      (ROWS*28*4)          // 1792   : relu(fc1) activations
#define OFF_B0P   (OFF_S + SZ_S)
#define SZ_B0P    (NPAD*4)
#define OFF_B1P   (OFF_B0P + SZ_B0P)
#define SZ_B1P    (NPAD*4)
#define OFF_WFC1  (OFF_B1P + SZ_B1P)
#define SZ_WFC1   (25*52*4)            // 5200
#define OFF_BFC1  (OFF_WFC1 + SZ_WFC1)
#define SZ_BFC1   (32*4)
#define OFF_WFC2  (OFF_BFC1 + SZ_BFC1)
#define SZ_WFC2   (32*4)
#define OFF_BFC2  (OFF_WFC2 + SZ_WFC2)
#define SZ_BFC2   16
#define SMEM_BYTES (OFF_BFC2 + SZ_BFC2) // 125920 B < 320KB WGP LDS

__device__ __forceinline__ __bf16 f2bf(float f) {
  union { float f; unsigned u; } x; x.f = f;
  unsigned r = x.u + 0x7FFFu + ((x.u >> 16) & 1u);   // round-to-nearest-even
  unsigned short h = (unsigned short)(r >> 16);
  return __builtin_bit_cast(__bf16, h);
}

__device__ __forceinline__ float sigm(float x) {
  return __builtin_amdgcn_rcpf(1.0f + __expf(-x));       // saturates cleanly
}
__device__ __forceinline__ float tanh_fast(float x) {
  float e = __expf(2.0f * x);
  return 1.0f - 2.0f * __builtin_amdgcn_rcpf(e + 1.0f);  // +/-1 at saturation
}

__device__ __forceinline__ v8f wmma_bf16(v16bf a, v16bf b, v8f c) {
  return __builtin_amdgcn_wmma_f32_16x16x32_bf16(false, a, false, b,
                                                 (short)0, c, false, false);
}

// Gate GEMM: gates[16 x 208] = A(16xK) * W(KxN) + bias.
// Each wave owns n-tiles {w, w+4, w+8, w+12} and processes them as pairs
// (nt, nt+4): two independent accumulator chains share one A fragment, so the
// scheduler can interleave WMMAs from the two chains into each other's
// D->C hazard windows (no NOPs between independent WMMAs) and A-side LDS
// traffic is halved.
// B fragments are pre-packed in LDS in the exact B-operand lane layout:
// frag(kt,nt): lane l holds 16 bf16 = K = kt*32 + (l<16?0:8) + {0..7, 16..23},
//              N = nt*16 + (l&15).
template<int NKT, int S>
__device__ __forceinline__ void gemm_gates(const __bf16* __restrict__ xh,
                                           const __bf16* __restrict__ Wf,
                                           const float*  __restrict__ bp,
                                           float* __restrict__ gates,
                                           int lane, int wave) {
  const int mcol  = lane & 15;
  const int koff  = (lane < 16) ? 0 : 8;   // A layout: lanes>=16 hold K runs +8
  const int mbase = (lane < 16) ? 0 : 8;   // D layout: lanes>=16 hold M rows +8
  for (int nt0 = wave; nt0 < NT; nt0 += 8) {       // pairs (w,w+4), (w+8,w+12)
    const int  nt1  = nt0 + 4;
    const bool has1 = (nt1 < NT);                  // wave-uniform
    float bv0 = bp[nt0*16 + mcol];
    v8f acc0 = {bv0,bv0,bv0,bv0,bv0,bv0,bv0,bv0};
    float bv1 = has1 ? bp[nt1*16 + mcol] : 0.f;
    v8f acc1 = {bv1,bv1,bv1,bv1,bv1,bv1,bv1,bv1};
#pragma unroll
    for (int kt = 0; kt < NKT; ++kt) {
      // A fragment (shared by both chains): two contiguous 8-element K runs
      const __bf16* rp = xh + mcol*S + kt*32 + koff;
      v8bf lo = *(const v8bf*)(rp);        // K = k0+koff    .. +7
      v8bf hi = *(const v8bf*)(rp + 16);   // K = k0+koff+16 .. +23
      v16bf a = __builtin_shufflevector(lo, hi, 0,1,2,3,4,5,6,7,8,9,10,11,12,13,14,15);
      v16bf b0 = *(const v16bf*)(Wf + (kt*NT + nt0)*512 + lane*16);
      acc0 = wmma_bf16(a, b0, acc0);
      if (has1) {
        v16bf b1 = *(const v16bf*)(Wf + (kt*NT + nt1)*512 + lane*16);
        acc1 = wmma_bf16(a, b1, acc1);
      }
    }
    {
      int n0 = nt0*16 + mcol;
#pragma unroll
      for (int r = 0; r < 8; ++r) gates[(mbase + r)*NPAD + n0] = acc0[r];
    }
    if (has1) {
      int n1 = nt1*16 + mcol;
#pragma unroll
      for (int r = 0; r < 8; ++r) gates[(mbase + r)*NPAD + n1] = acc1[r];
    }
  }
}

__global__ __launch_bounds__(NTHREADS)
void kozyra_lstm_scan_kernel(const float* __restrict__ feat,
                             const float* __restrict__ w_ih0, const float* __restrict__ w_hh0,
                             const float* __restrict__ b_0,
                             const float* __restrict__ w_ih1, const float* __restrict__ w_hh1,
                             const float* __restrict__ b_1,
                             const float* __restrict__ w_fc1, const float* __restrict__ b_fc1,
                             const float* __restrict__ w_fc2, const float* __restrict__ b_fc2,
                             float* __restrict__ out) {
  extern __shared__ unsigned char smem[];
  __bf16* W0f   = (__bf16*)(smem + OFF_W0F);
  __bf16* W1f   = (__bf16*)(smem + OFF_W1F);
  float*  gates = (float*) (smem + OFF_GATES);
  __bf16* xh0   = (__bf16*)(smem + OFF_XH0);
  __bf16* xh1   = (__bf16*)(smem + OFF_XH1);
  float*  h1f   = (float*) (smem + OFF_H1F);
  float*  sbuf  = (float*) (smem + OFF_S);
  float*  b0p   = (float*) (smem + OFF_B0P);
  float*  b1p   = (float*) (smem + OFF_B1P);
  float*  wfc1  = (float*) (smem + OFF_WFC1);
  float*  bfc1  = (float*) (smem + OFF_BFC1);
  float*  wfc2  = (float*) (smem + OFF_WFC2);
  float*  bfc2s = (float*) (smem + OFF_BFC2);

  const int tid  = threadIdx.x;
  const int lane = tid & 31;
  const int wave = tid >> 5;
  const int b0r  = blockIdx.x * ROWS;

  // ---------------- one-time setup (amortized over 128 steps) ----------------
  for (int i = tid; i < ROWS*K0PAD; i += NTHREADS) xh0[i] = f2bf(0.f);  // h0=delta=0
  for (int i = tid; i < ROWS*K1PAD; i += NTHREADS) xh1[i] = f2bf(0.f);  // h0=h1=0
  for (int i = tid; i < NPAD; i += NTHREADS) {
    b0p[i] = (i < G4) ? b_0[i] : 0.f;
    b1p[i] = (i < G4) ? b_1[i] : 0.f;
  }
  for (int i = tid; i < 25*50; i += NTHREADS) {
    int p = i / 50, j = i - p*50;
    wfc1[p*52 + j] = w_fc1[i];
  }
  if (tid < 25) { bfc1[tid] = b_fc1[tid]; wfc2[tid] = w_fc2[tid]; }
  if (tid == 0) bfc2s[0] = b_fc2[0];

  // Pack W0 = [w_ih0^T ; w_hh0^T] (K=67->96, N=200->208) into B-operand fragments.
  for (int i = tid; i < NKT0*NT*512; i += NTHREADS) {
    int frag = i >> 9, rem = i & 511;
    int fl = rem >> 4, v = rem & 15;
    int kt = frag / NT, nt = frag - kt*NT;
    int k = kt*32 + ((fl < 16) ? 0 : 8) + (v & 7) + ((v >> 3) << 4);
    int n = nt*16 + (fl & 15);
    float w = 0.f;
    if (n < G4) {
      if (k < 17)      w = w_ih0[n*17 + k];          // [x, prev_delta]
      else if (k < 67) w = w_hh0[n*50 + (k - 17)];   // h0 recurrent
    }
    W0f[i] = f2bf(w);
  }
  // Pack W1 = [w_ih1^T ; w_hh1^T] (K=100->128, N=200->208).
  for (int i = tid; i < NKT1*NT*512; i += NTHREADS) {
    int frag = i >> 9, rem = i & 511;
    int fl = rem >> 4, v = rem & 15;
    int kt = frag / NT, nt = frag - kt*NT;
    int k = kt*32 + ((fl < 16) ? 0 : 8) + (v & 7) + ((v >> 3) << 4);
    int n = nt*16 + (fl & 15);
    float w = 0.f;
    if (n < G4) {
      if (k < 50)       w = w_ih1[n*50 + k];          // h0 input
      else if (k < 100) w = w_hh1[n*50 + (k - 50)];   // h1 recurrent
    }
    W1f[i] = f2bf(w);
  }
  __syncthreads();

  float c0r[7] = {};   // cell state lives in registers: item (q,tid) -> (m,j) is static
  float c1r[7] = {};

#pragma unroll 1
  for (int t = 0; t < TSTEPS; ++t) {
    // ---- stream this step's features into xh0[:,0:16]; prefetch next step ----
#pragma unroll
    for (int q = 0; q < 2; ++q) {
      int e = tid + q*NTHREADS;            // 256 feature elements
      int m = e >> 4, d = e & 15;
      long base = (long)(b0r + m)*(TSTEPS*SD);
      xh0[m*K0PAD + d] = f2bf(feat[base + t*SD + d]);
      if (t + 1 < TSTEPS)
        __builtin_prefetch(&feat[base + (t+1)*SD + d], 0, 1);  // global_prefetch_b8
    }
    __syncthreads();

    // ---- layer0: gates = [x,delta,h0] @ W0 + b0  (WMMA bf16 -> f32) ----
    gemm_gates<NKT0, K0PAD>(xh0, W0f, b0p, gates, lane, wave);
    __syncthreads();

    // ---- layer0 cell update (f32 state, fast sigmoid/tanh) ----
#pragma unroll
    for (int q = 0; q < 7; ++q) {
      int idx = tid + q*NTHREADS;
      if (idx < ROWS*HID) {
        int m = idx / HID, j = idx - m*HID;
        const float* g = gates + m*NPAD;
        float ii = g[j], ff = g[HID+j], gg = g[2*HID+j], oo = g[3*HID+j];
        float c = sigm(ff)*c0r[q] + sigm(ii)*tanh_fast(gg);
        float h = sigm(oo)*tanh_fast(c);
        c0r[q] = c;
        __bf16 hb = f2bf(h);
        xh1[m*K1PAD + j]      = hb;   // layer1 input (this step)
        xh0[m*K0PAD + 17 + j] = hb;   // layer0 recurrent input (next step)
      }
    }
    __syncthreads();

    // ---- layer1: gates = [h0,h1] @ W1 + b1 ----
    gemm_gates<NKT1, K1PAD>(xh1, W1f, b1p, gates, lane, wave);
    __syncthreads();

    // ---- layer1 cell update ----
#pragma unroll
    for (int q = 0; q < 7; ++q) {
      int idx = tid + q*NTHREADS;
      if (idx < ROWS*HID) {
        int m = idx / HID, j = idx - m*HID;
        const float* g = gates + m*NPAD;
        float ii = g[j], ff = g[HID+j], gg = g[2*HID+j], oo = g[3*HID+j];
        float c = sigm(ff)*c1r[q] + sigm(ii)*tanh_fast(gg);
        float h = sigm(oo)*tanh_fast(c);
        c1r[q] = c;
        xh1[m*K1PAD + 50 + j] = f2bf(h);  // layer1 recurrent (next step)
        h1f[m*52 + j] = h;                // f32 copy for the FC head
      }
    }
    __syncthreads();

    // ---- fc1: relu(h1 @ w_fc1^T + b)  (16x25, K=50; VALU from LDS) ----
#pragma unroll
    for (int q = 0; q < 4; ++q) {
      int idx = tid + q*NTHREADS;
      if (idx < ROWS*25) {
        int m = idx / 25, p = idx - m*25;
        float acc = bfc1[p];
        const float* hr = h1f  + m*52;
        const float* wr = wfc1 + p*52;
        for (int j = 0; j < HID; ++j) acc = fmaf(hr[j], wr[j], acc);
        sbuf[m*28 + p] = fmaxf(acc, 0.f);
      }
    }
    __syncthreads();

    // ---- fc2 -> delta: write output and feed back into next-step input ----
    if (tid < ROWS) {
      int m = tid;
      float acc = bfc2s[0];
      for (int p = 0; p < 25; ++p) acc = fmaf(sbuf[m*28 + p], wfc2[p], acc);
      out[(long)(b0r + m)*TSTEPS + t] = acc;
      xh0[m*K0PAD + SD] = f2bf(acc);      // prev_delta slot (k=16)
    }
    __syncthreads();
  }
}

extern "C" void kernel_launch(void* const* d_in, const int* in_sizes, int n_in,
                              void* d_out, int out_size, void* d_ws, size_t ws_size,
                              hipStream_t stream) {
  (void)in_sizes; (void)n_in; (void)out_size; (void)d_ws; (void)ws_size;
  const float* feat  = (const float*)d_in[0];
  const float* w_ih0 = (const float*)d_in[1];
  const float* w_hh0 = (const float*)d_in[2];
  const float* b_0   = (const float*)d_in[3];
  const float* w_ih1 = (const float*)d_in[4];
  const float* w_hh1 = (const float*)d_in[5];
  const float* b_1   = (const float*)d_in[6];
  const float* w_fc1 = (const float*)d_in[7];
  const float* b_fc1 = (const float*)d_in[8];
  const float* w_fc2 = (const float*)d_in[9];
  const float* b_fc2 = (const float*)d_in[10];
  float* out = (float*)d_out;

  // Opt in to >64KB dynamic LDS (gfx1250 WGP has 320KB); deterministic, capture-safe.
  (void)hipFuncSetAttribute((const void*)kozyra_lstm_scan_kernel,
                            hipFuncAttributeMaxDynamicSharedMemorySize, SMEM_BYTES);

  dim3 grid(8192 / ROWS);   // 512 workgroups x 4 waves = 2048 waves
  dim3 block(NTHREADS);
  kozyra_lstm_scan_kernel<<<grid, block, SMEM_BYTES, stream>>>(
      feat, w_ih0, w_hh0, b_0, w_ih1, w_hh1, b_1, w_fc1, b_fc1, w_fc2, b_fc2, out);
}
